// LSTM_60292750901483
// MI455X (gfx1250) — matrix operand order repaired and checked by reading, not compiled, not run on hardware
//
#include <hip/hip_runtime.h>
#include <hip/hip_bf16.h>
#include <cstdint>

typedef __attribute__((ext_vector_type(16))) _Float16 v16h;
typedef __attribute__((ext_vector_type(8)))  _Float16 v8h;
typedef __attribute__((ext_vector_type(4)))  _Float16 v4h;
typedef __attribute__((ext_vector_type(8)))  float    v8f;
typedef __attribute__((ext_vector_type(4)))  unsigned int v4u;
typedef __attribute__((ext_vector_type(4)))  int      v4i;

#define TSEQ 512
#define DDIM 1024
#define HDIM 1024
#define NBLK 128               // persistent blocks
#define NTHR 256               // 8 waves / block

#if __has_builtin(__builtin_amdgcn_global_load_async_to_lds_b128) && \
    __has_builtin(__builtin_amdgcn_s_wait_asynccnt)
#define USE_ASYNC_LDS 1
typedef __attribute__((address_space(1))) v4i gbl_v4i;   // global b128 chunk
typedef __attribute__((address_space(3))) v4i lds_v4i;   // LDS b128 chunk
#endif

// ---------------------------------------------------------------------------
// K0: pack [Wx;Wh] (fp32, (K,4H) row-major) into per-lane WMMA B fragments, f16.
//   ntile = jt*4 + gate, original col n = gate*1024 + jt*16 + (lane&15)
//   fragment element j2 -> K = kc*32 + (lane>>4)*16 + j2 (contiguous per lane).
// ---------------------------------------------------------------------------
__global__ void lstm_pack_w(const float* __restrict__ Wx,
                            const float* __restrict__ Wh,
                            _Float16* __restrict__ Bp) {
    unsigned idx = blockIdx.x * 256u + threadIdx.x;     // 0 .. 8388607
    int j2    = idx & 15;
    int lane  = (idx >> 4) & 31;
    int kc    = (idx >> 9) & 63;
    int ntile = idx >> 15;                              // 0..255
    int k     = kc * 32 + ((lane >> 4) << 4) + j2;      // 0..2047
    int gate  = ntile & 3;
    int jt    = ntile >> 2;
    int n     = gate * 1024 + jt * 16 + (lane & 15);    // 0..4095
    float v = (k < DDIM) ? Wx[(size_t)k * 4096 + n]
                         : Wh[(size_t)(k - DDIM) * 4096 + n];
    Bp[idx] = (_Float16)v;
}

// ---------------------------------------------------------------------------
// K1: convert x fp32 -> f16, same (N,T,D) row-major layout. 4 elems / thread.
// ---------------------------------------------------------------------------
__global__ void lstm_cvt_x(const float* __restrict__ x,
                           _Float16* __restrict__ xh, int n4) {
    int i = blockIdx.x * 256 + threadIdx.x;
    if (i < n4) {
        float4 v = ((const float4*)x)[i];
        v4h o = { (_Float16)v.x, (_Float16)v.y, (_Float16)v.z, (_Float16)v.w };
        ((v4h*)xh)[i] = o;
    }
}

// ---------------------------------------------------------------------------
// K2: h0 -> f16 ping buffer 0; reset grid-barrier state.
// ---------------------------------------------------------------------------
__global__ void lstm_init(const float* __restrict__ h0,
                          _Float16* __restrict__ hbuf,
                          unsigned* __restrict__ bar) {
    int i = blockIdx.x * 256 + threadIdx.x;
    if (i < 64 * HDIM) hbuf[i] = (_Float16)h0[i];
    if (i == 0) { bar[0] = 0u; bar[1] = 0u; }
}

// ---------------------------------------------------------------------------
// Sense-reversing grid barrier (agent scope).
// ---------------------------------------------------------------------------
__device__ __forceinline__ void grid_barrier(unsigned* cnt, unsigned* gen,
                                             unsigned nblk) {
    __syncthreads();
    if (threadIdx.x == 0) {
        unsigned g = __hip_atomic_load(gen, __ATOMIC_RELAXED,
                                       __HIP_MEMORY_SCOPE_AGENT);
        unsigned a = __hip_atomic_fetch_add(cnt, 1u, __ATOMIC_ACQ_REL,
                                            __HIP_MEMORY_SCOPE_AGENT);
        if (a == nblk - 1) {
            __hip_atomic_store(cnt, 0u, __ATOMIC_RELAXED,
                               __HIP_MEMORY_SCOPE_AGENT);
            __hip_atomic_fetch_add(gen, 1u, __ATOMIC_RELEASE,
                                   __HIP_MEMORY_SCOPE_AGENT);
        } else {
            while (__hip_atomic_load(gen, __ATOMIC_ACQUIRE,
                                     __HIP_MEMORY_SCOPE_AGENT) == g)
                __builtin_amdgcn_s_sleep(2);
        }
    }
    __syncthreads();
}

// ---------------------------------------------------------------------------
// Gate activations via the CDNA5 hardware V_TANH_F32 when available.
// sigmoid(x) = 0.5 + 0.5*tanh(0.5*x)
// ---------------------------------------------------------------------------
__device__ __forceinline__ float fast_tanh(float x) {
#if __has_builtin(__builtin_amdgcn_tanhf)
    return __builtin_amdgcn_tanhf(x);
#elif __has_builtin(__builtin_amdgcn_tanh_f32)
    return __builtin_amdgcn_tanh_f32(x);
#else
    x = fminf(15.0f, fmaxf(-15.0f, x));          // branchless saturation
    float e = __expf(2.0f * x);
    return (e - 1.0f) / (e + 1.0f);
#endif
}
__device__ __forceinline__ float fast_sigmoid(float x) {
    return 0.5f + 0.5f * fast_tanh(0.5f * x);
}

// ---------------------------------------------------------------------------
// K3: persistent fused LSTM. 128 blocks x 256 threads (8 waves).
//   wave -> output tile: gid = blk*2 + (wave>>2); jt = gid&63; mtile = gid>>6;
//   gate = wave&3; ntile = jt*4 + gate.
// Per step: async-stage u=[x_t;h] (16 rows x 2048 f16, XOR-swizzled) into LDS,
// 64x v_wmma_f32_16x16x32_f16 over K=2048, intra-block gate fusion with
// c-state held in registers, one grid barrier per step.
// ---------------------------------------------------------------------------
__global__ void __launch_bounds__(NTHR)
lstm_persistent(const _Float16* __restrict__ xh,   // (64,512,1024) f16
                const _Float16* __restrict__ Bp,   // packed weights f16
                const float*    __restrict__ bias, // (4096,)
                _Float16*       __restrict__ hbuf, // 2 x (64*1024) f16 ping-pong
                float*          __restrict__ out,  // (64,512,1024) f32
                unsigned*       __restrict__ bar) {
    __shared__ __align__(16) unsigned char smem[65536];  // A-stage / acc overlay
    float* accs = (float*)smem;                          // 2*4*16*16 f32 = 8 KB

    const int tid    = threadIdx.x;
    const int lane   = tid & 31;
    const int wave   = tid >> 5;                 // 0..7
    const int grpL   = wave >> 2;                // 0..1
    const int gate   = wave & 3;                 // i,f,o,g
    const int gid    = blockIdx.x * 2 + grpL;    // 0..255
    const int jt     = gid & 63;
    const int ntile  = jt * 4 + gate;            // packed-weight col tile
    const int mtileB = blockIdx.x >> 5;          // block-uniform batch tile

    // gate-phase ownership (fixed for all steps) -> c stays in registers
    float cstate[2] = {0.0f, 0.0f};

    // t-invariant gate-phase constants (hoisted bias loads)
    int eGrp[2], eLoc[2];
    float bI[2], bF[2], bO[2], bG[2];
    size_t outBase[2];
    int hIdx[2];
    #pragma unroll
    for (int ee = 0; ee < 2; ++ee) {
        int e   = tid + ee * 256;
        int grp = e >> 8;
        int loc = e & 255;
        int m   = loc >> 4;
        int jj  = loc & 15;
        int jtE = (blockIdx.x * 2 + grp) & 63;
        int j   = jtE * 16 + jj;
        int bb  = mtileB * 16 + m;
        eGrp[ee] = grp; eLoc[ee] = loc;
        bI[ee] = bias[j];
        bF[ee] = bias[1024 + j];
        bO[ee] = bias[2048 + j];
        bG[ee] = bias[3072 + j];
        outBase[ee] = (size_t)bb * TSEQ * HDIM + j;
        hIdx[ee]    = bb * HDIM + j;
    }

    const int half = lane >> 4;                  // A-frag K-half select
    const int arow = lane & 15;                  // A-frag row (M)

    for (int t = 0; t < TSEQ; ++t) {
        const _Float16* hsrc = hbuf + (size_t)(t & 1) * (64 * HDIM);
        _Float16*       hdst = hbuf + (size_t)((t + 1) & 1) * (64 * HDIM);

        // ---- stage u = [x_t ; h_{t-1}] : 16 rows x 2048 f16, swizzled ----
        #pragma unroll
        for (int i = 0; i < 16; ++i) {
            int chunk = tid + i * 256;           // 0..4095 (16B chunks)
            int row   = chunk >> 8;              // 0..15
            int cc    = chunk & 255;             // 16B chunk within row
            int k0    = cc << 3;                 // element column
            const _Float16* src;
            if (k0 < DDIM)
                src = xh + ((size_t)(mtileB * 16 + row) * TSEQ + t) * DDIM + k0;
            else
                src = hsrc + (size_t)(mtileB * 16 + row) * HDIM + (k0 - DDIM);
            int scc = cc ^ row;                  // XOR swizzle (row<16)
            void* dst = smem + row * 4096 + scc * 16;
#if defined(USE_ASYNC_LDS)
            __builtin_amdgcn_global_load_async_to_lds_b128(
                (gbl_v4i*)src, (lds_v4i*)dst, 0, 0);
#else
            *(v4u*)dst = *(const v4u*)src;
#endif
        }
#if defined(USE_ASYNC_LDS)
        __builtin_amdgcn_s_wait_asynccnt(0);
#endif
        __syncthreads();

        // ---- GEMM: 64 x WMMA f16 16x16x32, K = 2048 ----
        v8f acc = {};
        const _Float16* bp =
            Bp + ((size_t)ntile * 64 * 32 + lane) * 16;   // + kc*512
        for (int kc = 0; kc < 64; ++kc) {
            int chunkA = kc * 4 + 2 * half;      // (kc*32 + 16*half)/8
            const v8h* a0p = (const v8h*)(smem + arow * 4096 + ((chunkA)     ^ arow) * 16);
            const v8h* a1p = (const v8h*)(smem + arow * 4096 + ((chunkA + 1) ^ arow) * 16);
            v8h alo = *a0p;
            v8h ahi = *a1p;
            v16h a;
            #pragma unroll
            for (int e = 0; e < 8; ++e) { a[e] = alo[e]; a[e + 8] = ahi[e]; }
            v16h b = *(const v16h*)(bp + (size_t)kc * 512);
            acc = __builtin_amdgcn_wmma_f32_16x16x32_f16(
                false, a, false, b, (short)0, acc, false, false);
        }
        __syncthreads();   // all waves done reading A-stage before overlay

        // ---- exchange accumulators through LDS: [grp][gate][m][jj] ----
        #pragma unroll
        for (int v = 0; v < 8; ++v) {
            int m  = (lane < 16) ? v : v + 8;
            int jj = lane & 15;
            accs[grpL * 1024 + gate * 256 + m * 16 + jj] = acc[v];
        }
        __syncthreads();

        // ---- fused gates + cell update: 2 elements / thread ----
        #pragma unroll
        for (int ee = 0; ee < 2; ++ee) {
            int grp = eGrp[ee], loc = eLoc[ee];
            float ai = accs[grp * 1024 + 0 * 256 + loc] + bI[ee];
            float af = accs[grp * 1024 + 1 * 256 + loc] + bF[ee];
            float ao = accs[grp * 1024 + 2 * 256 + loc] + bO[ee];
            float ag = accs[grp * 1024 + 3 * 256 + loc] + bG[ee];
            float iv = fast_sigmoid(ai);
            float fv = fast_sigmoid(af);
            float ov = fast_sigmoid(ao);
            float gv = fast_tanh(ag);
            float c  = fv * cstate[ee] + iv * gv;
            cstate[ee] = c;
            float h  = ov * fast_tanh(c);
            // streaming output: keep L2 for weights, not write-once data
            __builtin_nontemporal_store(h, &out[outBase[ee] + (size_t)t * HDIM]);
            hdst[hIdx[ee]] = (_Float16)h;
        }

        // ---- publish h_t to all blocks for step t+1 ----
        grid_barrier(bar, bar + 1, NBLK);
    }
}

// ---------------------------------------------------------------------------
// Host-side launch. Workspace layout (bytes):
//   [0, 16MiB)              packed weights f16 (4096 x 2048)
//   [16MiB, +64MiB)         x in f16 (64 x 512 x 1024)
//   [+...]                  h ping-pong f16 (2 x 64 x 1024)
//   [+...]                  barrier counters (2 x u32)
// ---------------------------------------------------------------------------
extern "C" void kernel_launch(void* const* d_in, const int* in_sizes, int n_in,
                              void* d_out, int out_size, void* d_ws, size_t ws_size,
                              hipStream_t stream) {
    const float* x    = (const float*)d_in[0];   // (64,512,1024)
    const float* h0   = (const float*)d_in[1];   // (64,1024)
    const float* Wx   = (const float*)d_in[2];   // (1024,4096)
    const float* Wh   = (const float*)d_in[3];   // (1024,4096)
    const float* bias = (const float*)d_in[4];   // (4096,)
    float* out = (float*)d_out;

    uint8_t* ws = (uint8_t*)d_ws;
    size_t off = 0;
    _Float16* Bp = (_Float16*)(ws + off); off += (size_t)4096 * 2048 * 2;
    _Float16* xh = (_Float16*)(ws + off); off += (size_t)64 * 512 * 1024 * 2;
    _Float16* hb = (_Float16*)(ws + off); off += (size_t)2 * 64 * 1024 * 2;
    unsigned* bar = (unsigned*)(ws + off);

    // K0: pack weights (8,388,608 fragment elements)
    lstm_pack_w<<<32768, 256, 0, stream>>>(Wx, Wh, Bp);
    // K1: convert x (33,554,432 elems, 4/thread)
    lstm_cvt_x<<<32768, 256, 0, stream>>>(x, xh, 64 * 512 * 1024 / 4);
    // K2: init h buffer + barrier
    lstm_init<<<256, 256, 0, stream>>>(h0, hb, bar);
    // K3: persistent fused LSTM
    lstm_persistent<<<NBLK, NTHR, 0, stream>>>(xh, Bp, bias, hb, out, bar);
}
